// EnsembleModel_111669150328
// MI455X (gfx1250) — compile-verified, matrix operand order checked by reading, not run on hardware
//
#include <hip/hip_runtime.h>
#include <math.h>

#define HIDN 256
#define EMBN 256
#define G4   1024
#define BB   8
#define LLEN 128
#define WWRD 48
#define SST  5
#define PPOS 32
#define NROW 1024

typedef __bf16 bf16x16 __attribute__((ext_vector_type(16)));
typedef float  v8f     __attribute__((ext_vector_type(8)));
typedef unsigned int u32x4 __attribute__((ext_vector_type(4)));

// Global (address-space 1) pointer types: keep weight loads on the
// global_load path (LOADcnt only, saddr+voffset form) even after laundering.
typedef const __attribute__((address_space(1))) unsigned short* gwptr;
typedef const __attribute__((address_space(1))) u32x4*          gqptr;

union Frag16 {
  bf16x16 v;
  unsigned short s[16];
  u32x4 q[2];
};

static __device__ __forceinline__ unsigned short f2bf(float x){
  union { float f; unsigned u; } u; u.f = x;
  unsigned r = (u.u + 0x7FFFu + ((u.u >> 16) & 1u)) >> 16;
  return (unsigned short)r;
}

static __device__ __forceinline__ float sigm(float x){ return 1.f / (1.f + __expf(-x)); }

static __device__ __forceinline__ v8f wmma_bf16(Frag16 a, Frag16 b, v8f c){
  return __builtin_amdgcn_wmma_f32_16x16x32_bf16(false, a.v, false, b.v, (short)0, c, false, false);
}

// Launder a pointer so LICM cannot hoist loads based on it out of a loop.
#define LAUNDER_PTR(p) asm volatile("" : "+s"(p))

// k-local index within a 32-wide K tile for element e, lane-half khalf
static __device__ __forceinline__ int kloc(int khalf, int e){
  int g = e >> 1, p = e & 1;
  return (g < 4) ? (8*khalf + 2*g + p) : (16 + 8*khalf + 2*(g-4) + p);
}

// A fragment from LDS bf16 row-major [16][stride] (lowers to 2x ds_load_b128)
static __device__ __forceinline__ Frag16 load_a(const unsigned short* A, int stride,
                                                int kt, int sub, int khalf){
  Frag16 f;
  const unsigned short* row = A + sub * stride + kt * 32;
#pragma unroll
  for (int g = 0; g < 8; ++g){
    int kl = (g < 4) ? (8*khalf + 2*g) : (16 + 8*khalf + 2*(g-4));
    unsigned pr = *(const unsigned*)(row + kl);
    f.s[2*g]   = (unsigned short)(pr & 0xffffu);
    f.s[2*g+1] = (unsigned short)(pr >> 16);
  }
  return f;
}

// B fragment from pre-swizzled global weights (lane-major, 16 contiguous bf16 / lane)
static __device__ __forceinline__ Frag16 load_b(gwptr tile, int lane){
  Frag16 f;
  gqptr p = (gqptr)(tile + lane * 16);
  f.q[0] = p[0];
  f.q[1] = p[1];
  return f;
}

// Convert W[O][K] f32 -> bf16 pre-swizzled fragment order: ((kt*NT+nt)*32+lane)*16+e
__global__ void swizzle_kernel(const float* __restrict__ W, unsigned short* __restrict__ dst,
                               int O, int K){
  int total = O * K;
  for (int d = blockIdx.x*blockDim.x + threadIdx.x; d < total; d += gridDim.x*blockDim.x){
    int e = d & 15;
    int lane = (d >> 4) & 31;
    int tile = d >> 9;
    int NT = O >> 4;
    int kt = tile / NT;
    int nt = tile - kt*NT;
    int sub = lane & 15, khalf = lane >> 4;
    int k = kt*32 + kloc(khalf, e);
    int o = nt*16 + sub;
    dst[d] = f2bf(W[o*K + k]);
  }
}

// Word-level LSTM + fused attention (online softmax).
// 64 blocks x 16 rows; 512 threads = 16 waves; wave w owns hidden cols [w*16, w*16+16).
__global__ __launch_bounds__(512)
void word_kernel(const int* __restrict__ tok, const float* __restrict__ emb,
                 const unsigned short* __restrict__ wih_sw,
                 const unsigned short* __restrict__ whh_sw,
                 const float* __restrict__ bias,
                 const unsigned short* __restrict__ ws1_sw,
                 const float* __restrict__ ws2,
                 float* __restrict__ att){
  __shared__ unsigned short xA[16*EMBN];
  __shared__ unsigned short hA[16*HIDN];
  __shared__ float lpart[16];
  __shared__ float rmax[16], rsum[16], sc0[16], sc1[16];
  __shared__ int toks[16];

  const int tid = threadIdx.x;
  const int wave = tid >> 5, lane = tid & 31;
  const int sub = lane & 15, khalf = lane >> 4;
  const int rbase = blockIdx.x * 16;
  const int col = wave*16 + sub;       // this lane's output column
  const v8f z8 = {0.f,0.f,0.f,0.f,0.f,0.f,0.f,0.f};

  for (int i = tid; i < 16*HIDN; i += 512) hA[i] = 0;
  if (tid < 16){ rmax[tid] = -3.0e38f; rsum[tid] = 0.f; lpart[tid] = 0.f; }
  v8f c = z8, attacc = z8;
  __syncthreads();

  for (int t = 0; t < WWRD; ++t){
    // Launder weight pointers (global AS): loads stay inside the loop, no spills
    gwptr wih_p = (gwptr)wih_sw;
    gwptr whh_p = (gwptr)whh_sw;
    gwptr ws1_p = (gwptr)ws1_sw;
    LAUNDER_PTR(wih_p);
    LAUNDER_PTR(whh_p);
    LAUNDER_PTR(ws1_p);

    if (tid < 16) toks[tid] = tok[(rbase + tid)*WWRD + t];
    __syncthreads();
    for (int idx = tid; idx < 16*EMBN; idx += 512){
      int m = idx >> 8, j = idx & 255;
      xA[idx] = f2bf(emb[(long)toks[m]*EMBN + j]);
    }
    __syncthreads();

    // gates: acc[q] for q in {i,f,g,o}, this wave's 16 columns
    v8f acc[4] = {z8, z8, z8, z8};
    for (int kt = 0; kt < EMBN/32; ++kt){
      Frag16 af = load_a(xA, EMBN, kt, sub, khalf);
      Frag16 b0 = load_b(wih_p + ((kt*64 + 0*16 + wave) << 9), lane);
      Frag16 b1 = load_b(wih_p + ((kt*64 + 1*16 + wave) << 9), lane);
      Frag16 b2 = load_b(wih_p + ((kt*64 + 2*16 + wave) << 9), lane);
      Frag16 b3 = load_b(wih_p + ((kt*64 + 3*16 + wave) << 9), lane);
      acc[0] = wmma_bf16(af, b0, acc[0]);
      acc[1] = wmma_bf16(af, b1, acc[1]);
      acc[2] = wmma_bf16(af, b2, acc[2]);
      acc[3] = wmma_bf16(af, b3, acc[3]);
    }
    for (int kt = 0; kt < HIDN/32; ++kt){
      Frag16 af = load_a(hA, HIDN, kt, sub, khalf);
      Frag16 b0 = load_b(whh_p + ((kt*64 + 0*16 + wave) << 9), lane);
      Frag16 b1 = load_b(whh_p + ((kt*64 + 1*16 + wave) << 9), lane);
      Frag16 b2 = load_b(whh_p + ((kt*64 + 2*16 + wave) << 9), lane);
      Frag16 b3 = load_b(whh_p + ((kt*64 + 3*16 + wave) << 9), lane);
      acc[0] = wmma_bf16(af, b0, acc[0]);
      acc[1] = wmma_bf16(af, b1, acc[1]);
      acc[2] = wmma_bf16(af, b2, acc[2]);
      acc[3] = wmma_bf16(af, b3, acc[3]);
    }
    __syncthreads();

    v8f hr;
    {
      float bi = bias[col], bff = bias[HIDN+col], bg = bias[2*HIDN+col], bo = bias[3*HIDN+col];
#pragma unroll
      for (int r = 0; r < 8; ++r){
        float iv = sigm(acc[0][r] + bi);
        float fv = sigm(acc[1][r] + bff);
        float gv = tanhf(acc[2][r] + bg);
        float ov = sigm(acc[3][r] + bo);
        float cv = fv * c[r] + iv * gv;
        c[r] = cv;
        float hv = ov * tanhf(cv);
        hr[r] = hv;
        int m = r + 8*khalf;
        hA[m*HIDN + col] = f2bf(hv);
      }
    }
    __syncthreads();

    // hbar = tanh(h @ ws1.T); per-row logit partials
    v8f a2 = z8;
    for (int kt = 0; kt < HIDN/32; ++kt){
      Frag16 af = load_a(hA, HIDN, kt, sub, khalf);
      Frag16 bf = load_b(ws1_p + ((kt*16 + wave) << 9), lane);
      a2 = wmma_bf16(af, bf, a2);
    }
    {
      float w2 = ws2[col];
#pragma unroll
      for (int r = 0; r < 8; ++r){
        float v = tanhf(a2[r]) * w2;
        v += __shfl_xor(v, 1, 32);
        v += __shfl_xor(v, 2, 32);
        v += __shfl_xor(v, 4, 32);
        v += __shfl_xor(v, 8, 32);
        if (sub == 0) atomicAdd(&lpart[r + 8*khalf], v);
      }
    }
    __syncthreads();
    if (tid < 16){
      float l = lpart[tid];
      lpart[tid] = 0.f;
      if (toks[tid] == 0) l -= 10000.f;
      float mo = rmax[tid];
      float mn = fmaxf(mo, l);
      float c0 = __expf(mo - mn);
      float c1 = __expf(l - mn);
      rsum[tid] = rsum[tid]*c0 + c1;
      rmax[tid] = mn;
      sc0[tid] = c0; sc1[tid] = c1;
    }
    __syncthreads();
#pragma unroll
    for (int r = 0; r < 8; ++r){
      int m = r + 8*khalf;
      attacc[r] = attacc[r]*sc0[m] + sc1[m]*hr[r];
    }
  }

#pragma unroll
  for (int r = 0; r < 8; ++r){
    int m = r + 8*khalf;
    att[(rbase + m)*HIDN + col] = attacc[r] / rsum[m];
  }
}

// Generic WMMA LSTM: conv (nvalid=8, T=128, perm=null) and session (nvalid=32, T=32, perm)
__global__ __launch_bounds__(512)
void lstm_kernel(const unsigned short* __restrict__ wih_sw,
                 const unsigned short* __restrict__ whh_sw,
                 const float* __restrict__ bias, const float* __restrict__ xsrc,
                 const int* __restrict__ perm, float* __restrict__ out,
                 int nvalid, int T){
  __shared__ unsigned short xA[16*HIDN];
  __shared__ unsigned short hA[16*HIDN];
  const int tid = threadIdx.x;
  const int wave = tid >> 5, lane = tid & 31;
  const int sub = lane & 15, khalf = lane >> 4;
  const int rbase = blockIdx.x * 16;
  const int col = wave*16 + sub;
  const v8f z8 = {0.f,0.f,0.f,0.f,0.f,0.f,0.f,0.f};

  for (int i = tid; i < 16*HIDN; i += 512) hA[i] = 0;
  v8f c = z8;
  __syncthreads();

  for (int t = 0; t < T; ++t){
    gwptr wih_p = (gwptr)wih_sw;
    gwptr whh_p = (gwptr)whh_sw;
    LAUNDER_PTR(wih_p);
    LAUNDER_PTR(whh_p);

    for (int idx = tid; idx < 16*HIDN; idx += 512){
      int m = idx >> 8, j = idx & 255;
      int grow = rbase + m;
      float v = 0.f;
      if (grow < nvalid){
        int sr = perm ? perm[grow*T + t] : (grow*T + t);
        v = xsrc[sr*HIDN + j];
      }
      xA[idx] = f2bf(v);
    }
    __syncthreads();

    v8f acc[4] = {z8, z8, z8, z8};
    for (int kt = 0; kt < HIDN/32; ++kt){
      Frag16 af = load_a(xA, HIDN, kt, sub, khalf);
      Frag16 b0 = load_b(wih_p + ((kt*64 + 0*16 + wave) << 9), lane);
      Frag16 b1 = load_b(wih_p + ((kt*64 + 1*16 + wave) << 9), lane);
      Frag16 b2 = load_b(wih_p + ((kt*64 + 2*16 + wave) << 9), lane);
      Frag16 b3 = load_b(wih_p + ((kt*64 + 3*16 + wave) << 9), lane);
      acc[0] = wmma_bf16(af, b0, acc[0]);
      acc[1] = wmma_bf16(af, b1, acc[1]);
      acc[2] = wmma_bf16(af, b2, acc[2]);
      acc[3] = wmma_bf16(af, b3, acc[3]);
    }
    for (int kt = 0; kt < HIDN/32; ++kt){
      Frag16 af = load_a(hA, HIDN, kt, sub, khalf);
      Frag16 b0 = load_b(whh_p + ((kt*64 + 0*16 + wave) << 9), lane);
      Frag16 b1 = load_b(whh_p + ((kt*64 + 1*16 + wave) << 9), lane);
      Frag16 b2 = load_b(whh_p + ((kt*64 + 2*16 + wave) << 9), lane);
      Frag16 b3 = load_b(whh_p + ((kt*64 + 3*16 + wave) << 9), lane);
      acc[0] = wmma_bf16(af, b0, acc[0]);
      acc[1] = wmma_bf16(af, b1, acc[1]);
      acc[2] = wmma_bf16(af, b2, acc[2]);
      acc[3] = wmma_bf16(af, b3, acc[3]);
    }
    __syncthreads();

    {
      float bi = bias[col], bff = bias[HIDN+col], bg = bias[2*HIDN+col], bo = bias[3*HIDN+col];
#pragma unroll
      for (int r = 0; r < 8; ++r){
        float iv = sigm(acc[0][r] + bi);
        float fv = sigm(acc[1][r] + bff);
        float gv = tanhf(acc[2][r] + bg);
        float ov = sigm(acc[3][r] + bo);
        float cv = fv * c[r] + iv * gv;
        c[r] = cv;
        float hv = ov * tanhf(cv);
        int m = r + 8*khalf;
        hA[m*HIDN + col] = f2bf(hv);
        int grow = rbase + m;
        if (grow < nvalid) out[(grow*T + t)*HIDN + col] = hv;
      }
    }
    __syncthreads();
  }
}

// Sequential state-transition scan over L steps
__global__ __launch_bounds__(256)
void scan_kernel(const int* __restrict__ stm, const float* __restrict__ conv,
                 const float* __restrict__ sess, const float* __restrict__ Wp,
                 const float* __restrict__ bp, float* __restrict__ smat){
  __shared__ float prev[BB][SST][HIDN];
  __shared__ float oneres[BB][HIDN];
  __shared__ int st[BB*SST];
  const int tid = threadIdx.x;
  const int j = tid;
  for (int i = tid; i < BB*SST*HIDN; i += 256) ((float*)prev)[i] = 0.f;
  __syncthreads();

  for (int t = 0; t < LLEN; ++t){
    if (tid < BB*SST){
      int b = tid / SST, s = tid - b*SST;
      st[tid] = stm[(b*LLEN + t)*SST + s];
    }
    __syncthreads();

    float vg[BB][SST-1];
#pragma unroll
    for (int b = 0; b < BB; ++b){
      float sum = 0.f;
#pragma unroll
      for (int s = 0; s < SST; ++s){
        int sv = st[b*SST + s];
        float g = 0.f, v = 0.f;
        if (sv > 0){
          int posn = sv - 1; if (posn > PPOS-1) posn = PPOS-1;
          int sidx = (s == 0) ? 0 : (s-1);
          g = sess[((b*(SST-1) + sidx)*PPOS + posn)*HIDN + j];
          v = g;
        } else if (sv == -1){
          v = prev[b][s][j];
        }
        if (s > 0) vg[b][s-1] = (sv > 0) ? g : 0.f;
        if (sv != 0) sum += v;
      }
      oneres[b][j] = sum * 0.25f;
    }
    __syncthreads();

#pragma unroll
    for (int b = 0; b < BB; ++b)
#pragma unroll
      for (int s = 1; s < SST; ++s){
        float v = vg[b][s-1];
        prev[b][s][j] = v;
        smat[((b*LLEN + t)*SST + s)*HIDN + j] = v;
      }

    int tcs = (t == 0) ? 0 : (t-1);
#pragma unroll
    for (int b = 0; b < BB; ++b){
      float a = bp[j];
      const float* wrow = Wp + j*(2*HIDN);
      const float* orow = oneres[b];
      const float* crow = conv + (b*LLEN + tcs)*HIDN;
      for (int k = 0; k < HIDN; ++k) a += orow[k]*wrow[k];
      for (int k = 0; k < HIDN; ++k) a += crow[k]*wrow[HIDN + k];
      float v = a > 0.f ? a : 0.f;
      prev[b][0][j] = v;
      smat[((b*LLEN + t)*SST + 0)*HIDN + j] = v;
    }
    __syncthreads();
  }
}

// up = relu([att,conv] @ Ws.T + bs); scores = state_matrix . up; log_softmax
__global__ __launch_bounds__(512)
void final_kernel(const unsigned short* __restrict__ ws_sw, const float* __restrict__ bs,
                  const float* __restrict__ att, const float* __restrict__ conv,
                  const float* __restrict__ smat, float* __restrict__ out){
  __shared__ unsigned short aA[16*512];
  __shared__ float up[16][HIDN];
  __shared__ float sc[16][SST];
  const int tid = threadIdx.x;
  const int wave = tid >> 5, lane = tid & 31;
  const int sub = lane & 15, khalf = lane >> 4;
  const int rbase = blockIdx.x * 16;
  const int col = wave*16 + sub;
  const v8f z8 = {0.f,0.f,0.f,0.f,0.f,0.f,0.f,0.f};

  for (int idx = tid; idx < 16*512; idx += 512){
    int m = idx >> 9, j = idx & 511;
    int n = rbase + m;
    float v = (j < HIDN) ? att[n*HIDN + j] : conv[n*HIDN + (j - HIDN)];
    aA[idx] = f2bf(v);
  }
  __syncthreads();

  gwptr wsp = (gwptr)ws_sw;
  v8f a2 = z8;
  for (int kt = 0; kt < 16; ++kt){
    Frag16 af = load_a(aA, 512, kt, sub, khalf);
    Frag16 bf = load_b(wsp + ((kt*16 + wave) << 9), lane);
    a2 = wmma_bf16(af, bf, a2);
  }
  {
    float bb = bs[col];
#pragma unroll
    for (int r = 0; r < 8; ++r){
      int m = r + 8*khalf;
      float v = a2[r] + bb;
      up[m][col] = v > 0.f ? v : 0.f;
    }
  }
  __syncthreads();

  if (tid < 16*SST){
    int m = tid / SST, s = tid - m*SST;
    int n = rbase + m;
    const float* srow = smat + (n*SST + s)*HIDN;
    float d = 0.f;
    for (int k = 0; k < HIDN; ++k) d += srow[k]*up[m][k];
    sc[m][s] = d;
  }
  __syncthreads();

  if (tid < 16){
    int n = rbase + tid;
    float mx = sc[tid][0];
    for (int s = 1; s < SST; ++s) mx = fmaxf(mx, sc[tid][s]);
    float se = 0.f;
    for (int s = 0; s < SST; ++s) se += __expf(sc[tid][s] - mx);
    float lse = __logf(se) + mx;
    for (int s = 0; s < SST; ++s) out[n*SST + s] = sc[tid][s] - lse;
  }
}

extern "C" void kernel_launch(void* const* d_in, const int* in_sizes, int n_in,
                              void* d_out, int out_size, void* d_ws, size_t ws_size,
                              hipStream_t stream){
  (void)in_sizes; (void)n_in; (void)out_size; (void)ws_size;
  const int*   tok  = (const int*)d_in[0];
  const int*   perm = (const int*)d_in[2];
  const int*   stm  = (const int*)d_in[3];
  const float* emb  = (const float*)d_in[5];
  const float* uWih = (const float*)d_in[6];
  const float* uWhh = (const float*)d_in[7];
  const float* ub   = (const float*)d_in[8];
  const float* ws1  = (const float*)d_in[9];
  const float* ws2  = (const float*)d_in[10];
  const float* cWih = (const float*)d_in[11];
  const float* cWhh = (const float*)d_in[12];
  const float* cb   = (const float*)d_in[13];
  const float* sWih = (const float*)d_in[14];
  const float* sWhh = (const float*)d_in[15];
  const float* sb   = (const float*)d_in[16];
  const float* Wp   = (const float*)d_in[17];
  const float* bp   = (const float*)d_in[18];
  const float* Wsm  = (const float*)d_in[19];
  const float* bs   = (const float*)d_in[20];
  float* out = (float*)d_out;

  char* wsb = (char*)d_ws;
  size_t off = 0;
  auto alloc = [&](size_t bytes)->char*{
    char* p = wsb + off; off += (bytes + 255) & ~(size_t)255; return p;
  };
  unsigned short* uwih_sw = (unsigned short*)alloc((size_t)G4*EMBN*2);
  unsigned short* uwhh_sw = (unsigned short*)alloc((size_t)G4*HIDN*2);
  unsigned short* ws1_sw  = (unsigned short*)alloc((size_t)HIDN*HIDN*2);
  unsigned short* cwih_sw = (unsigned short*)alloc((size_t)G4*HIDN*2);
  unsigned short* cwhh_sw = (unsigned short*)alloc((size_t)G4*HIDN*2);
  unsigned short* swih_sw = (unsigned short*)alloc((size_t)G4*HIDN*2);
  unsigned short* swhh_sw = (unsigned short*)alloc((size_t)G4*HIDN*2);
  unsigned short* wsm_sw  = (unsigned short*)alloc((size_t)HIDN*2*HIDN*2);
  float* att   = (float*)alloc((size_t)NROW*HIDN*4);
  float* convb = (float*)alloc((size_t)BB*LLEN*HIDN*4);
  float* sessb = (float*)alloc((size_t)32*PPOS*HIDN*4);
  float* smat  = (float*)alloc((size_t)BB*LLEN*SST*HIDN*4);

  swizzle_kernel<<<64, 256, 0, stream>>>(uWih, uwih_sw, G4, EMBN);
  swizzle_kernel<<<64, 256, 0, stream>>>(uWhh, uwhh_sw, G4, HIDN);
  swizzle_kernel<<<64, 256, 0, stream>>>(ws1,  ws1_sw,  HIDN, HIDN);
  swizzle_kernel<<<64, 256, 0, stream>>>(cWih, cwih_sw, G4, HIDN);
  swizzle_kernel<<<64, 256, 0, stream>>>(cWhh, cwhh_sw, G4, HIDN);
  swizzle_kernel<<<64, 256, 0, stream>>>(sWih, swih_sw, G4, HIDN);
  swizzle_kernel<<<64, 256, 0, stream>>>(sWhh, swhh_sw, G4, HIDN);
  swizzle_kernel<<<64, 256, 0, stream>>>(Wsm,  wsm_sw,  HIDN, 2*HIDN);

  word_kernel<<<64, 512, 0, stream>>>(tok, emb, uwih_sw, uwhh_sw, ub, ws1_sw, ws2, att);
  lstm_kernel<<<1, 512, 0, stream>>>(cwih_sw, cwhh_sw, cb, att, nullptr, convb, 8, LLEN);
  lstm_kernel<<<2, 512, 0, stream>>>(swih_sw, swhh_sw, sb, att, perm, sessb, 32, PPOS);
  scan_kernel<<<1, 256, 0, stream>>>(stm, convb, sessb, Wp, bp, smat);
  final_kernel<<<64, 512, 0, stream>>>(wsm_sw, bs, att, convb, smat, out);
}
